// ConvGSCSNN_15831249453645
// MI455X (gfx1250) — matrix operand order repaired and checked by reading, not compiled, not run on hardware
//
#include <hip/hip_runtime.h>

// ---------------------------------------------------------------------------
// CDNA5 (gfx1250) SNN pipeline:
//   conv1 (scalar, scatters f16 im2col) -> conv2 via WMMA -> fc1 via WMMA
//   -> 101-step recurrent scan via WMMA.
// ---------------------------------------------------------------------------

typedef __attribute__((ext_vector_type(16))) _Float16 v16h;
typedef __attribute__((ext_vector_type(8)))  _Float16 h8;
typedef __attribute__((ext_vector_type(8)))  float    v8f;

#define BN_EPS 1e-5f

__device__ __forceinline__ v8f zero8() {
    v8f z;
#pragma unroll
    for (int i = 0; i < 8; ++i) z[i] = 0.f;
    return z;
}

// Build a 16-element f16 fragment from two 8-halfword (16B-aligned) loads.
__device__ __forceinline__ v16h frag_from(const _Float16* plo, const _Float16* phi) {
    h8 lo = *(const h8*)plo;
    h8 hi = *(const h8*)phi;
    return __builtin_shufflevector(lo, hi, 0, 1, 2, 3, 4, 5, 6, 7,
                                           8, 9, 10, 11, 12, 13, 14, 15);
}

// ---------------------------------------------------------------------------
// Kernel 0: weight precision prep (f32 -> f16), pad fc_out (12x256 -> 16x256)
// ---------------------------------------------------------------------------
__global__ void prep_weights_k(const float* __restrict__ fc1_w,
                               const float* __restrict__ rec_w,
                               const float* __restrict__ out_w,
                               const float* __restrict__ w2,
                               _Float16* __restrict__ fc1_h,
                               _Float16* __restrict__ rec_h,
                               _Float16* __restrict__ out_h,
                               _Float16* __restrict__ w2_h) {
    int i = blockIdx.x * blockDim.x + threadIdx.x;
    if (i < 256 * 512) fc1_h[i] = (_Float16)fc1_w[i];
    if (i < 256 * 256) rec_h[i] = (_Float16)rec_w[i];
    if (i < 64 * 96) w2_h[i] = (_Float16)w2[i];
    if (i < 16 * 256) {
        int n = i >> 8, k = i & 255;
        out_h[i] = (n < 12) ? (_Float16)out_w[n * 256 + k] : (_Float16)0.f;
    }
}

// ---------------------------------------------------------------------------
// Kernel 1: one 256-thread WG per 16 consecutive frames.
//   Phase A: stage x (16x120 f32) in LDS.
//   Phase B: conv1+BN+ReLU+pool (scalar), scatter pooled values directly into
//            the f16 im2col A-matrix Ai[(f*16+p)*96 + ci*3 + k].
//   Phase C: conv2 as WMMA GEMM (M=256, K=96, N=64); BN+ReLU+avgpool applied
//            on the accumulator fragment; result -> flats[f][cout*8+pp] (f16).
//   Phase D: fc1 as WMMA GEMM (16 x 512) @ (512 x 256) -> I_ff (T,B,256) f32.
// ---------------------------------------------------------------------------
__global__ __launch_bounds__(256) void conv_fc1_k(
    const float* __restrict__ x,
    const float* __restrict__ w1, const float* __restrict__ g1,
    const float* __restrict__ b1, const float* __restrict__ m1,
    const float* __restrict__ vv1,
    const _Float16* __restrict__ w2h, const float* __restrict__ g2,
    const float* __restrict__ b2, const float* __restrict__ m2,
    const float* __restrict__ vv2,
    const _Float16* __restrict__ fc1_h,
    float* __restrict__ Iff) {
    __shared__ __align__(16) float    xs[16 * 120];     //  7.7 KB
    __shared__ __align__(16) _Float16 Ai[256 * 96];     // 48.0 KB im2col
    __shared__ __align__(16) _Float16 flats[16 * 512];  // 16.4 KB

    const int tid = threadIdx.x;
    const int frame0 = blockIdx.x * 16;

    // ---- Phase A: load 16 frames (contiguous 1920 floats) ----
    for (int i = tid; i < 16 * 120; i += 256)
        xs[i] = x[(size_t)frame0 * 120 + i];
    __syncthreads();

    // ---- Phase B: conv1 (32,3,5) + BN + ReLU + avgpool2, scatter to im2col ----
    for (int e = tid; e < 16 * 576; e += 256) {
        int f = e / 576;
        int r = e - f * 576;
        int ci = r / 18;       // conv1 out channel == conv2 in channel
        int q = r - ci * 18;   // pooled position, 0..17
        float s0 = 0.f, s1 = 0.f;
        const float* xf = &xs[f * 120];
#pragma unroll
        for (int c = 0; c < 3; ++c) {
            const float* xr = xf + c * 40 + 2 * q;
            const float* wr = w1 + (ci * 3 + c) * 5;
#pragma unroll
            for (int k = 0; k < 5; ++k) {
                s0 += xr[k] * wr[k];
                s1 += xr[k + 1] * wr[k];
            }
        }
        float inv = g1[ci] * rsqrtf(vv1[ci] + BN_EPS);
        float sh = b1[ci] - m1[ci] * inv;
        float a0 = fmaxf(s0 * inv + sh, 0.f);
        float a1 = fmaxf(s1 * inv + sh, 0.f);
        _Float16 hv = (_Float16)(0.5f * (a0 + a1));
        // im2col scatter: value c1[f][ci][q] -> A[(f*16+p)*96 + ci*3 + k], p=q-k
#pragma unroll
        for (int k = 0; k < 3; ++k) {
            int p = q - k;
            if (p >= 0 && p < 16)
                Ai[(f * 16 + p) * 96 + ci * 3 + k] = hv;
        }
    }
    __syncthreads();

    const int lane = tid & 31;
    const int wave = tid >> 5;
    const int laneLo = lane & 15;
    const int laneHi = lane >> 4;

    // ---- Phase C: conv2 as WMMA (per wave: 2 frames x 4 cout-tiles) ----
    {
        const int f0 = wave * 2;
        const int f1 = f0 + 1;
        float inv2[4], sh2[4];
#pragma unroll
        for (int nt = 0; nt < 4; ++nt) {
            int co = nt * 16 + laneLo;
            inv2[nt] = g2[co] * rsqrtf(vv2[co] + BN_EPS);
            sh2[nt] = b2[co] - m2[co] * inv2[nt];
        }
        v8f c20[4], c21[4];
#pragma unroll
        for (int nt = 0; nt < 4; ++nt) { c20[nt] = zero8(); c21[nt] = zero8(); }

        const _Float16* A0base = &Ai[(f0 * 16 + laneLo) * 96];
        const _Float16* A1base = &Ai[(f1 * 16 + laneLo) * 96];
#pragma unroll
        for (int ks = 0; ks < 3; ++ks) {
            int k0 = ks * 32;
            v16h A0 = frag_from(A0base + k0 + laneHi * 16,
                                A0base + k0 + laneHi * 16 + 8);
            v16h A1 = frag_from(A1base + k0 + laneHi * 16,
                                A1base + k0 + laneHi * 16 + 8);
#pragma unroll
            for (int nt = 0; nt < 4; ++nt) {
                const _Float16* Bb = w2h + (size_t)(nt * 16 + laneLo) * 96
                                     + k0 + laneHi * 16;
                v16h Bf = frag_from(Bb, Bb + 8);
                c20[nt] = __builtin_amdgcn_wmma_f32_16x16x32_f16(
                    false, A0, false, Bf, (short)0, c20[nt], false, false);
                c21[nt] = __builtin_amdgcn_wmma_f32_16x16x32_f16(
                    false, A1, false, Bf, (short)0, c21[nt], false, false);
            }
        }
        // BN + ReLU + avgpool2 on accumulator; pos = laneHi*8 + r, pool pairs
        // are adjacent r slots of the same lane. pooled idx = laneHi*4 + j.
#pragma unroll
        for (int nt = 0; nt < 4; ++nt) {
            int co = nt * 16 + laneLo;
#pragma unroll
            for (int j = 0; j < 4; ++j) {
                float a0 = fmaxf(c20[nt][2 * j] * inv2[nt] + sh2[nt], 0.f);
                float a1 = fmaxf(c20[nt][2 * j + 1] * inv2[nt] + sh2[nt], 0.f);
                flats[f0 * 512 + co * 8 + laneHi * 4 + j] =
                    (_Float16)(0.5f * (a0 + a1));
                float b0v = fmaxf(c21[nt][2 * j] * inv2[nt] + sh2[nt], 0.f);
                float b1v = fmaxf(c21[nt][2 * j + 1] * inv2[nt] + sh2[nt], 0.f);
                flats[f1 * 512 + co * 8 + laneHi * 4 + j] =
                    (_Float16)(0.5f * (b0v + b1v));
            }
        }
    }
    __syncthreads();

    // ---- Phase D: fc1 (16 x 512) @ (512 x 256) via WMMA ----
    const int n0 = wave * 32 + laneLo;
    const int n1 = n0 + 16;
    const _Float16* Arow = &flats[laneLo * 512];
    const _Float16* B0row = fc1_h + (size_t)n0 * 512;
    const _Float16* B1row = fc1_h + (size_t)n1 * 512;

    v8f acc0 = zero8(), acc1 = zero8();
#pragma unroll 4
    for (int ks = 0; ks < 16; ++ks) {
        int k0 = ks * 32;
        v16h A = frag_from(Arow + k0 + laneHi * 16,
                           Arow + k0 + laneHi * 16 + 8);
        v16h B0 = frag_from(B0row + k0 + laneHi * 16,
                            B0row + k0 + laneHi * 16 + 8);
        acc0 = __builtin_amdgcn_wmma_f32_16x16x32_f16(false, A, false, B0,
                                                      (short)0, acc0, false, false);
        v16h B1 = frag_from(B1row + k0 + laneHi * 16,
                            B1row + k0 + laneHi * 16 + 8);
        acc1 = __builtin_amdgcn_wmma_f32_16x16x32_f16(false, A, false, B1,
                                                      (short)0, acc1, false, false);
    }

    // Store I_ff in (T, B, 256) layout: row m -> frame -> (b, t)
#pragma unroll
    for (int r = 0; r < 8; ++r) {
        int m = laneHi * 8 + r;
        unsigned frame = (unsigned)(frame0 + m);
        unsigned b = frame / 101u;
        unsigned t = frame - b * 101u;
        size_t base = ((size_t)t * 1024u + b) * 256u;
        Iff[base + (unsigned)(wave * 32) + laneLo] = acc0[r];
        Iff[base + (unsigned)(wave * 32) + 16u + laneLo] = acc1[r];
    }
}

// ---------------------------------------------------------------------------
// Kernel 2: 101-step SNN scan. Each WG owns 16 batch rows; spike tile lives in
// LDS (f16) and feeds the recurrent WMMA each step. LIF state is held in the
// same lane/VGPR mapping as the WMMA accumulator, so updates are register-local.
// Wave 0 additionally runs the padded fc_out WMMA tile and the vout leak.
// ---------------------------------------------------------------------------
__global__ __launch_bounds__(256) void snn_scan_k(
    const float* __restrict__ Iff,
    const _Float16* __restrict__ rec_h,
    const _Float16* __restrict__ out_h,
    const float* __restrict__ alpha,
    const float* __restrict__ rho,
    const float* __restrict__ beta_a,
    const float* __restrict__ beta_out,
    float* __restrict__ out) {
    __shared__ __align__(16) _Float16 spks[16 * 256];  // 8 KB spike tile

    const int tid = threadIdx.x;
    const int b0 = blockIdx.x * 16;
    const int lane = tid & 31;
    const int wave = tid >> 5;
    const int laneLo = lane & 15;
    const int laneHi = lane >> 4;

    for (int i = tid; i < 16 * 256; i += 256) spks[i] = (_Float16)0.f;

    const int nA = wave * 32 + laneLo;  // n column, tile A
    const int nB = nA + 16;             // n column, tile B
    const float alA = alpha[nA], alB = alpha[nB];
    const float rhA = rho[nA], rhB = rho[nB];
    const float baA = beta_a[nA], baB = beta_a[nB];
    const float bout = (laneLo < 12) ? beta_out[laneLo] : 0.f;

    v8f vA = zero8(), vB = zero8();
    v8f aA = zero8(), aB = zero8();
    v8f sA = zero8(), sB = zero8();
    v8f vout = zero8(), vsum = zero8();

    const _Float16* RA = rec_h + (size_t)nA * 256;
    const _Float16* RB = rec_h + (size_t)nB * 256;
    const _Float16* RO = out_h + (size_t)laneLo * 256;

    __syncthreads();

    for (int t = 0; t < 101; ++t) {
        // ---- recurrent drive: spk_prev (16x256, LDS) @ W_rec^T (256x256) ----
        v8f dA = zero8(), dB = zero8();
#pragma unroll
        for (int ks = 0; ks < 8; ++ks) {
            int k0 = ks * 32;
            v16h A = frag_from(&spks[laneLo * 256 + k0 + laneHi * 16],
                               &spks[laneLo * 256 + k0 + laneHi * 16 + 8]);
            v16h Wa = frag_from(RA + k0 + laneHi * 16,
                                RA + k0 + laneHi * 16 + 8);
            dA = __builtin_amdgcn_wmma_f32_16x16x32_f16(false, A, false, Wa,
                                                        (short)0, dA, false, false);
            v16h Wb = frag_from(RB + k0 + laneHi * 16,
                                RB + k0 + laneHi * 16 + 8);
            dB = __builtin_amdgcn_wmma_f32_16x16x32_f16(false, A, false, Wb,
                                                        (short)0, dB, false, false);
        }
        __syncthreads();  // all spk_prev reads done before overwrite

        // ---- elementwise LIF / adaptation update, emit spk_new to LDS ----
        const float* Irow = Iff + ((size_t)t * 1024 + (size_t)b0) * 256;
#pragma unroll
        for (int r = 0; r < 8; ++r) {
            int m = laneHi * 8 + r;
            {
                float I = dA[r] + Irow[m * 256 + nA];
                float an = rhA * aA[r] + (1.f - rhA) * sA[r];
                float vn = alA * (vA[r] - sA[r]) + (1.f - alA) * I - baA * an;
                float sn = (vn > 1.f) ? 1.f : 0.f;
                aA[r] = an; vA[r] = vn; sA[r] = sn;
                spks[m * 256 + nA] = (_Float16)sn;
            }
            {
                float I = dB[r] + Irow[m * 256 + nB];
                float an = rhB * aB[r] + (1.f - rhB) * sB[r];
                float vn = alB * (vB[r] - sB[r]) + (1.f - alB) * I - baB * an;
                float sn = (vn > 1.f) ? 1.f : 0.f;
                aB[r] = an; vB[r] = vn; sB[r] = sn;
                spks[m * 256 + nB] = (_Float16)sn;
            }
        }
        __syncthreads();  // spk_new visible to all waves

        // ---- wave 0: vout = beta*vout + (1-beta)*(spk_new @ W_out^T) ----
        if (wave == 0) {
            v8f so = zero8();
#pragma unroll
            for (int ks = 0; ks < 8; ++ks) {
                int k0 = ks * 32;
                v16h A = frag_from(&spks[laneLo * 256 + k0 + laneHi * 16],
                                   &spks[laneLo * 256 + k0 + laneHi * 16 + 8]);
                v16h Wo = frag_from(RO + k0 + laneHi * 16,
                                    RO + k0 + laneHi * 16 + 8);
                so = __builtin_amdgcn_wmma_f32_16x16x32_f16(false, A, false, Wo,
                                                            (short)0, so, false, false);
            }
#pragma unroll
            for (int r = 0; r < 8; ++r) {
                vout[r] = bout * vout[r] + (1.f - bout) * so[r];
                vsum[r] += vout[r];
            }
        }
        // next iteration's first barrier orders wave0's reads vs. new writes
    }

    if (wave == 0 && laneLo < 12) {
#pragma unroll
        for (int r = 0; r < 8; ++r) {
            int m = laneHi * 8 + r;
            out[(b0 + m) * 12 + laneLo] = vsum[r] * (1.f / 101.f);
        }
    }
}

// ---------------------------------------------------------------------------
// Host launcher
// ---------------------------------------------------------------------------
extern "C" void kernel_launch(void* const* d_in, const int* in_sizes, int n_in,
                              void* d_out, int out_size, void* d_ws, size_t ws_size,
                              hipStream_t stream) {
    const float* x = (const float*)d_in[0];
    const float* w1 = (const float*)d_in[1];
    const float* g1 = (const float*)d_in[2];
    const float* b1 = (const float*)d_in[3];
    const float* m1 = (const float*)d_in[4];
    const float* v1 = (const float*)d_in[5];
    const float* w2 = (const float*)d_in[6];
    const float* g2 = (const float*)d_in[7];
    const float* b2 = (const float*)d_in[8];
    const float* m2 = (const float*)d_in[9];
    const float* v2 = (const float*)d_in[10];
    const float* fc1_w = (const float*)d_in[11];
    const float* rec_w = (const float*)d_in[12];
    const float* out_w = (const float*)d_in[13];
    const float* alpha = (const float*)d_in[14];
    const float* rho = (const float*)d_in[15];
    const float* beta_a = (const float*)d_in[16];
    const float* beta_o = (const float*)d_in[17];

    // Workspace layout (bytes):
    //   [0, 262144)            fc1_w  as f16 (256x512)
    //   [262144, 393216)       rec_w  as f16 (256x256)
    //   [393216, 401408)       fc_out as f16 padded (16x256)
    //   [401408, 413696)       conv2_w as f16 (64x96)
    //   [413696, +105.9MB)     I_ff   f32 (T=101, B=1024, 256)
    char* ws = (char*)d_ws;
    _Float16* fc1_h = (_Float16*)ws;
    _Float16* rec_h = (_Float16*)(ws + 262144);
    _Float16* out_h = (_Float16*)(ws + 393216);
    _Float16* w2_h = (_Float16*)(ws + 401408);
    float* Iff = (float*)(ws + 413696);

    prep_weights_k<<<512, 256, 0, stream>>>(fc1_w, rec_w, out_w, w2,
                                            fc1_h, rec_h, out_h, w2_h);
    conv_fc1_k<<<6464, 256, 0, stream>>>(x, w1, g1, b1, m1, v1,
                                         w2_h, g2, b2, m2, v2, fc1_h, Iff);
    snn_scan_k<<<64, 256, 0, stream>>>(Iff, rec_h, out_h,
                                       alpha, rho, beta_a, beta_o, (float*)d_out);
}
